// ABCNN_67843303407731
// MI455X (gfx1250) — matrix operand order repaired and checked by reading, not compiled, not run on hardware
//
#include <hip/hip_runtime.h>
#include <hip/hip_bf16.h>
#include <math.h>

// ---------------------------------------------------------------- constants
#define BB 512
#define SS 128
#define EE 300
#define EP 320          // K-padded E for bf16 wmma operands (multiple of 32)
#define LL 2
#define HH 300
#define N_SE (BB*SS*EE) // 19,660,800

typedef __attribute__((ext_vector_type(16))) __bf16 v16bf;
typedef __attribute__((ext_vector_type(8)))  __bf16 v8bf;
typedef __attribute__((ext_vector_type(8)))  float  v8f;

// ------------------------------------------------------------- frag loaders
// A-matrix 16x32 bf16: lane&15 = M row; lanes 0-15 hold K=0..15,
// lanes 16-31 hold K=16..31 (contiguous, matching B)
__device__ __forceinline__ v16bf load_frag_a(const __bf16* rowptr, int kstep, int half) {
    union { v16bf v; v8bf h[2]; } u;
    const v8bf* p = reinterpret_cast<const v8bf*>(rowptr + kstep * 32 + half * 16);
    u.h[0] = p[0];   // K = kstep*32 + half*16 + {0..7}
    u.h[1] = p[1];   // K = kstep*32 + half*16 + 8 + {0..7}
    return u.v;
}
// B-matrix 32x16 bf16: lane&15 = N col; lanes 0-15 hold K=0..15,
// lanes 16-31 hold K=16..31; dword r packs k=2r,2r+1 -> contiguous 16 values.
__device__ __forceinline__ v16bf load_frag_b(const __bf16* rowptr, int kstep, int half) {
    return *reinterpret_cast<const v16bf*>(rowptr + kstep * 32 + half * 16);
}

// --------------------------------------------- async global->LDS slab copy
// 16 bytes per lane per issue; nchunk = bytes/16. Tracked by ASYNCcnt.
__device__ __forceinline__ void async_copy_slab(const __bf16* __restrict__ g,
                                                unsigned lds_base, int nchunk,
                                                int tid, int nthreads) {
    for (int c = tid; c < nchunk; c += nthreads) {
        unsigned laddr = lds_base + (unsigned)c * 16u;
        const __bf16* gp = g + (size_t)c * 8;
        asm volatile("global_load_async_to_lds_b128 %0, %1, off"
                     :: "v"(laddr), "v"(gp) : "memory");
    }
}
__device__ __forceinline__ void async_wait_all() {
    asm volatile("s_wait_asynccnt 0" ::: "memory");
}

// ---------------------------------------------------------------- kernels
__global__ void gather_kernel(const int* __restrict__ seq, const float* __restrict__ emb,
                              float* __restrict__ s) {
    int idx = blockIdx.x * 256 + threadIdx.x;
    if (idx >= N_SE) return;
    int e = idx % EE, bs = idx / EE;
    int tok = seq[bs];
    s[idx] = emb[(size_t)tok * EE + e];
}

// out[b*stride + e] = mean over S
__global__ void mean_kernel(const float* __restrict__ x, float* __restrict__ out, int stride) {
    int idx = blockIdx.x * 256 + threadIdx.x;
    if (idx >= BB * EE) return;
    int b = idx / EE, e = idx % EE;
    float acc = 0.f;
    for (int s = 0; s < SS; ++s) acc += x[((size_t)(b * SS + s)) * EE + e];
    out[(size_t)b * stride + e] = acc * (1.0f / SS);
}

// W[l][j][e] (f32) -> Wt[l][e][j] (bf16), e zero-padded to EP
__global__ void prepw_kernel(const float* __restrict__ W, __bf16* __restrict__ Wt) {
    int idx = blockIdx.x * 256 + threadIdx.x;
    if (idx >= LL * EP * SS) return;
    int j = idx & (SS - 1);
    int e = (idx >> 7) % EP;
    int l = idx / (EP * SS);
    float v = (e < EE) ? W[(size_t)l * SS * EE + (size_t)j * EE + e] : 0.f;
    Wt[idx] = (__bf16)v;
}

// mask-by-valid, f32->bf16 with K pad, and row sum-of-squares
__global__ void __launch_bounds__(128) pack_kernel(const float* __restrict__ x,
                                                   const int* __restrict__ seq,
                                                   __bf16* __restrict__ xh,
                                                   float* __restrict__ nrm) {
    int bs = blockIdx.x;                       // b*S + s
    float v = (seq[bs] != 0) ? 1.0f : 0.0f;
    const float* rowp = x + (size_t)bs * EE;
    __bf16* outp = xh + (size_t)bs * EP;
    int t = threadIdx.x;
    float ss = 0.f;
    for (int e = t; e < EP; e += 128) {
        float val = (e < EE) ? rowp[e] * v : 0.f;
        outp[e] = (__bf16)val;
        ss += val * val;
    }
    __shared__ float red[128];
    red[t] = ss; __syncthreads();
    for (int off = 64; off > 0; off >>= 1) {
        if (t < off) red[t] += red[t + off];
        __syncthreads();
    }
    if (t == 0) nrm[bs] = red[0];
}

// ----------------------------------------------------------- match kernels
// One workgroup per batch element. Stage m1[b], m2[b] (80KB bf16 each) into
// LDS with async copies; wave w owns row-block ti=w, keeps its A fragments in
// VGPRs, sweeps all 8 tj tiles with B fragments from LDS.
// MODE 0: store A and A^T (bf16).  MODE 1: row/col sums of A2 via shfl+atomics.
template <int MODE>
__global__ void __launch_bounds__(256) match_stage_kernel(
        const __bf16* __restrict__ m1, const __bf16* __restrict__ m2,
        const float* __restrict__ n1, const float* __restrict__ n2,
        __bf16* __restrict__ Ah, __bf16* __restrict__ Ath,
        float* __restrict__ w1, float* __restrict__ w2) {
    __shared__ __bf16 sm1[SS * EP];   // 80 KB
    __shared__ __bf16 sm2[SS * EP];   // 80 KB
    int b = blockIdx.x;
    int tid = threadIdx.x;
    unsigned l1 = (unsigned)(size_t)&sm1[0];
    unsigned l2 = (unsigned)(size_t)&sm2[0];
    async_copy_slab(m1 + (size_t)b * SS * EP, l1, SS * EP * 2 / 16, tid, 256);
    async_copy_slab(m2 + (size_t)b * SS * EP, l2, SS * EP * 2 / 16, tid, 256);
    async_wait_all();
    __syncthreads();

    int ti = tid >> 5;                 // wave id = row-block
    int lane = tid & 31;
    int row = lane & 15, half = lane >> 4;

    const __bf16* arow = sm1 + (size_t)(ti * 16 + row) * EP;
    v16bf afrag[EP / 32];
#pragma unroll
    for (int k = 0; k < EP / 32; ++k) afrag[k] = load_frag_a(arow, k, half);

    float n1v[8];
#pragma unroll
    for (int r = 0; r < 8; ++r) n1v[r] = n1[b * SS + ti * 16 + r + half * 8];

    for (int tj = 0; tj < 8; ++tj) {
        const __bf16* brow = sm2 + (size_t)(tj * 16 + row) * EP;
        v8f c = {};
#pragma unroll
        for (int k = 0; k < EP / 32; ++k) {
            v16bf bb = load_frag_b(brow, k, half);
            c = __builtin_amdgcn_wmma_f32_16x16x32_bf16(false, afrag[k], false, bb,
                                                        (short)0, c, false, false);
        }
        int j = tj * 16 + row;
        float nn2 = n2[b * SS + j];
        if (MODE == 0) {
#pragma unroll
            for (int r = 0; r < 8; ++r) {
                int i = ti * 16 + r + half * 8;
                float d2 = n1v[r] + nn2 - 2.0f * c[r];
                float d = sqrtf(fmaxf(d2, 0.0f));
                __bf16 av = (__bf16)(1.0f / (1.0f + d));
                Ah [((size_t)(b * SS + i)) * SS + j] = av;
                Ath[((size_t)(b * SS + j)) * SS + i] = av;
            }
        } else {
            float av[8];
            float colsum = 0.f;
#pragma unroll
            for (int r = 0; r < 8; ++r) {
                float d2 = n1v[r] + nn2 - 2.0f * c[r];
                float d = sqrtf(fmaxf(d2, 0.0f));
                av[r] = 1.0f / (1.0f + d);
                colsum += av[r];
            }
            atomicAdd(&w2[b * SS + j], colsum);   // col sums over this row-block
#pragma unroll
            for (int r = 0; r < 8; ++r) {          // row sums over 16 cols
                float v = av[r];
                for (int m = 1; m < 16; m <<= 1) v += __shfl_xor(v, m, 32);
                if (row == 0) atomicAdd(&w1[b * SS + ti * 16 + r + half * 8], v);
            }
        }
    }
}

// f[b,i,e] = sum_j Ah[b,i,j] * Wt[e,j].  One workgroup per b: stage Ah[b]
// (32KB) + Wt slab (80KB) in LDS; wave = ti, A frags in VGPRs, sweep 19 e-tiles.
__global__ void __launch_bounds__(256) awgemm_kernel(const __bf16* __restrict__ Ah,
                                                     const __bf16* __restrict__ Wt,
                                                     float* __restrict__ f) {
    __shared__ __bf16 sA[SS * SS];    // 32 KB
    __shared__ __bf16 sW[EP * SS];    // 80 KB
    int b = blockIdx.x;
    int tid = threadIdx.x;
    unsigned la = (unsigned)(size_t)&sA[0];
    unsigned lw = (unsigned)(size_t)&sW[0];
    async_copy_slab(Ah + (size_t)b * SS * SS, la, SS * SS * 2 / 16, tid, 256);
    async_copy_slab(Wt, lw, EP * SS * 2 / 16, tid, 256);
    async_wait_all();
    __syncthreads();

    int ti = tid >> 5;
    int lane = tid & 31;
    int row = lane & 15, half = lane >> 4;

    const __bf16* arow = sA + (size_t)(ti * 16 + row) * SS;
    v16bf afrag[SS / 32];
#pragma unroll
    for (int k = 0; k < SS / 32; ++k) afrag[k] = load_frag_a(arow, k, half);

    for (int te = 0; te < 19; ++te) {
        const __bf16* brow = sW + (size_t)(te * 16 + row) * SS;
        v8f c = {};
#pragma unroll
        for (int k = 0; k < SS / 32; ++k) {
            v16bf bb = load_frag_b(brow, k, half);
            c = __builtin_amdgcn_wmma_f32_16x16x32_bf16(false, afrag[k], false, bb,
                                                        (short)0, c, false, false);
        }
        int e = te * 16 + row;
        if (e < EE) {
#pragma unroll
            for (int r = 0; r < 8; ++r) {
                int i = ti * 16 + r + half * 8;
                f[((size_t)(b * SS + i)) * EE + e] = c[r];
            }
        }
    }
}

// 2-channel 3x3 conv over (S,E), pad 1, + bias, tanh
__global__ void conv_tanh_kernel(const float* __restrict__ s_in,
                                 const float* __restrict__ f_in,
                                 const float* __restrict__ cw,
                                 const float* __restrict__ cb,
                                 float* __restrict__ o) {
    int idx = blockIdx.x * 256 + threadIdx.x;
    if (idx >= N_SE) return;
    int e = idx % EE, bs = idx / EE;
    int s = bs & (SS - 1), b = bs >> 7;
    float acc = cb[0];
    const float* src = s_in + (size_t)(b * SS) * EE;
#pragma unroll
    for (int c = 0; c < 2; ++c) {
#pragma unroll
        for (int dy = -1; dy <= 1; ++dy) {
            int ss2 = s + dy;
            if (ss2 < 0 || ss2 >= SS) continue;
#pragma unroll
            for (int dx = -1; dx <= 1; ++dx) {
                int ee = e + dx;
                if (ee < 0 || ee >= EE) continue;
                acc += cw[c * 9 + (dy + 1) * 3 + (dx + 1)] * src[(size_t)ss2 * EE + ee];
            }
        }
        src = f_in + (size_t)(b * SS) * EE;
    }
    o[idx] = tanhf(acc);
}

__global__ void zero_kernel(float* __restrict__ p, int n) {
    int idx = blockIdx.x * 256 + threadIdx.x;
    if (idx < n) p[idx] = 0.f;
}

// s += avg3(o * w)  (kernel-3 mean pool over seq, count_include_pad)
__global__ void update_kernel(float* __restrict__ s1, const float* __restrict__ o,
                              const float* __restrict__ w) {
    int idx = blockIdx.x * 256 + threadIdx.x;
    if (idx >= N_SE) return;
    int e = idx % EE, bs = idx / EE;
    int s = bs & (SS - 1), b = bs >> 7;
    float acc = 0.f;
#pragma unroll
    for (int dy = -1; dy <= 1; ++dy) {
        int t = s + dy;
        if (t < 0 || t >= SS) continue;
        acc += o[((size_t)(b * SS + t)) * EE + e] * w[b * SS + t];
    }
    s1[idx] += acc * (1.0f / 3.0f);
}

// fc1 -> layernorm -> relu -> fc2 -> softmax; one block per batch row
__global__ void __launch_bounds__(512) head_kernel(const float* __restrict__ res1,
                                                   const float* __restrict__ res2,
                                                   const float* __restrict__ fc1w,
                                                   const float* __restrict__ fc1b,
                                                   const float* __restrict__ lng,
                                                   const float* __restrict__ lnb,
                                                   const float* __restrict__ fc2w,
                                                   const float* __restrict__ fc2b,
                                                   float* __restrict__ out) {
    __shared__ float xs[1800];
    __shared__ float hs[HH];
    __shared__ float red[512];
    __shared__ float sc[4];
    int b = blockIdx.x, t = threadIdx.x;
    for (int k = t; k < 1800; k += 512)
        xs[k] = (k < 900) ? res1[(size_t)b * 900 + k] : res2[(size_t)b * 900 + k - 900];
    __syncthreads();
    if (t < HH) {
        float h = fc1b[t];
        const float* wr = fc1w + (size_t)t * 1800;
        for (int k = 0; k < 1800; ++k) h += xs[k] * wr[k];
        hs[t] = h;
    }
    __syncthreads();
    red[t] = (t < HH) ? hs[t] : 0.f; __syncthreads();
    for (int off = 256; off > 0; off >>= 1) { if (t < off) red[t] += red[t + off]; __syncthreads(); }
    if (t == 0) sc[0] = red[0] * (1.0f / HH);
    __syncthreads();
    float mu = sc[0];
    float dv = (t < HH) ? (hs[t] - mu) : 0.f;
    red[t] = dv * dv; __syncthreads();
    for (int off = 256; off > 0; off >>= 1) { if (t < off) red[t] += red[t + off]; __syncthreads(); }
    if (t == 0) sc[1] = red[0] * (1.0f / HH);
    __syncthreads();
    float var = sc[1];
    float hn = 0.f;
    if (t < HH) {
        hn = (hs[t] - mu) * rsqrtf(var + 1e-5f) * lng[t] + lnb[t];
        hn = fmaxf(hn, 0.f);
    }
    red[t] = (t < HH) ? hn * fc2w[t] : 0.f; __syncthreads();
    for (int off = 256; off > 0; off >>= 1) { if (t < off) red[t] += red[t + off]; __syncthreads(); }
    if (t == 0) sc[2] = red[0] + fc2b[0];
    __syncthreads();
    red[t] = (t < HH) ? hn * fc2w[HH + t] : 0.f; __syncthreads();
    for (int off = 256; off > 0; off >>= 1) { if (t < off) red[t] += red[t + off]; __syncthreads(); }
    if (t == 0) {
        float o0 = sc[2], o1 = red[0] + fc2b[1];
        out[b * 2] = o0; out[b * 2 + 1] = o1;
        float m = fmaxf(o0, o1);
        float e0 = expf(o0 - m), e1 = expf(o1 - m);
        float z = e0 + e1;
        out[BB * 2 + b * 2]     = e0 / z;
        out[BB * 2 + b * 2 + 1] = e1 / z;
    }
}

// ---------------------------------------------------------------- launch
extern "C" void kernel_launch(void* const* d_in, const int* in_sizes, int n_in,
                              void* d_out, int out_size, void* d_ws, size_t ws_size,
                              hipStream_t stream) {
    const int*   seq1   = (const int*)  d_in[0];
    const int*   seq2   = (const int*)  d_in[1];
    const float* emb    = (const float*)d_in[2];
    const float* W      = (const float*)d_in[3];
    const float* conv_w = (const float*)d_in[4];
    const float* conv_b = (const float*)d_in[5];
    const float* fc1w   = (const float*)d_in[6];
    const float* fc1b   = (const float*)d_in[7];
    const float* lng    = (const float*)d_in[8];
    const float* lnb    = (const float*)d_in[9];
    const float* fc2w   = (const float*)d_in[10];
    const float* fc2b   = (const float*)d_in[11];

    char* ws = (char*)d_ws;
    size_t off = 0;
    auto alloc = [&](size_t bytes) -> void* {
        void* p = ws + off;
        off += (bytes + 255) & ~(size_t)255;
        return p;
    };
    float*  s1   = (float*) alloc((size_t)N_SE * 4);
    float*  s2   = (float*) alloc((size_t)N_SE * 4);
    float*  fbuf = (float*) alloc((size_t)N_SE * 4);
    float*  o1   = (float*) alloc((size_t)N_SE * 4);
    float*  o2   = (float*) alloc((size_t)N_SE * 4);
    __bf16* h1   = (__bf16*)alloc((size_t)BB * SS * EP * 2);
    __bf16* h2   = (__bf16*)alloc((size_t)BB * SS * EP * 2);
    __bf16* Ah   = (__bf16*)alloc((size_t)BB * SS * SS * 2);
    __bf16* Ath  = (__bf16*)alloc((size_t)BB * SS * SS * 2);
    __bf16* Wth  = (__bf16*)alloc((size_t)LL * EP * SS * 2);
    float*  n1   = (float*) alloc((size_t)BB * SS * 4);
    float*  n2   = (float*) alloc((size_t)BB * SS * 4);
    float*  w1   = (float*) alloc((size_t)BB * SS * 4);
    float*  w2   = (float*) alloc((size_t)BB * SS * 4);
    float*  res1 = (float*) alloc((size_t)BB * 900 * 4);
    float*  res2 = (float*) alloc((size_t)BB * 900 * 4);

    const int gElem = (N_SE + 255) / 256;           // 76800
    const int gMean = (BB * EE + 255) / 256;        // 600

    gather_kernel<<<gElem, 256, 0, stream>>>(seq1, emb, s1);
    gather_kernel<<<gElem, 256, 0, stream>>>(seq2, emb, s2);
    prepw_kernel<<<(LL * EP * SS + 255) / 256, 256, 0, stream>>>(W, Wth);
    mean_kernel<<<gMean, 256, 0, stream>>>(s1, res1 + 0, 900);
    mean_kernel<<<gMean, 256, 0, stream>>>(s2, res2 + 0, 900);

    for (int l = 0; l < LL; ++l) {
        pack_kernel<<<BB * SS, 128, 0, stream>>>(s1, seq1, h1, n1);
        pack_kernel<<<BB * SS, 128, 0, stream>>>(s2, seq2, h2, n2);
        match_stage_kernel<0><<<BB, 256, 0, stream>>>(h1, h2, n1, n2, Ah, Ath,
                                                      nullptr, nullptr);

        awgemm_kernel<<<BB, 256, 0, stream>>>(Ah, Wth + (size_t)l * EP * SS, fbuf);
        conv_tanh_kernel<<<gElem, 256, 0, stream>>>(s1, fbuf, conv_w + l * 18, conv_b + l, o1);
        awgemm_kernel<<<BB, 256, 0, stream>>>(Ath, Wth + (size_t)l * EP * SS, fbuf);
        conv_tanh_kernel<<<gElem, 256, 0, stream>>>(s2, fbuf, conv_w + l * 18, conv_b + l, o2);

        mean_kernel<<<gMean, 256, 0, stream>>>(o1, res1 + (l + 1) * EE, 900);
        mean_kernel<<<gMean, 256, 0, stream>>>(o2, res2 + (l + 1) * EE, 900);

        pack_kernel<<<BB * SS, 128, 0, stream>>>(o1, seq1, h1, n1);
        pack_kernel<<<BB * SS, 128, 0, stream>>>(o2, seq2, h2, n2);
        zero_kernel<<<(BB * SS + 255) / 256, 256, 0, stream>>>(w1, BB * SS);
        zero_kernel<<<(BB * SS + 255) / 256, 256, 0, stream>>>(w2, BB * SS);
        match_stage_kernel<1><<<BB, 256, 0, stream>>>(h1, h2, n1, n2, nullptr, nullptr,
                                                      w1, w2);

        update_kernel<<<gElem, 256, 0, stream>>>(s1, o1, w1);
        update_kernel<<<gElem, 256, 0, stream>>>(s2, o2, w2);
    }

    head_kernel<<<BB, 512, 0, stream>>>(res1, res2, fc1w, fc1b, lng, lnb,
                                        fc2w, fc2b, (float*)d_out);
}
